// NeRFRenderer_46411416600911
// MI455X (gfx1250) — matrix operand order repaired and checked by reading, not compiled, not run on hardware
//
#include <hip/hip_runtime.h>

// CDNA5 / gfx1250 NeRF renderer (wave32, WMMA-based MLP eval).
// ws layout: [ sigma0: 16384*64 f32 | z_all: 16384*128 f32 ]  (~12.6 MB)

typedef __attribute__((ext_vector_type(2)))  float    v2f;
typedef __attribute__((ext_vector_type(8)))  float    v8f;
typedef __attribute__((ext_vector_type(16))) _Float16 v16h;

#define N_RAYS 16384
#define HID    64
#define NSTEP  64
#define TFINE  128

__device__ __forceinline__ float softplus_f(float x) {
  return x > 0.f ? x + log1pf(expf(-x)) : log1pf(expf(x));
}
__device__ __forceinline__ float sigmoid_f(float x) {
  return 1.f / (1.f + expf(-x));
}
__device__ __forceinline__ float clamp1(float v) { return fminf(fmaxf(v, -1.f), 1.f); }

__device__ __forceinline__ void ray_near_far(float ox, float oy, float oz,
                                             float dx, float dy, float dz,
                                             float& nearv, float& farv) {
  float t0, t1, lo, hi, nv, fv;
  t0 = (-1.f - ox) / (dx + 1e-15f); t1 = (1.f - ox) / (dx + 1e-15f);
  nv = fminf(t0, t1); fv = fmaxf(t0, t1);
  t0 = (-1.f - oy) / (dy + 1e-15f); t1 = (1.f - oy) / (dy + 1e-15f);
  lo = fminf(t0, t1); hi = fmaxf(t0, t1); nv = fmaxf(nv, lo); fv = fminf(fv, hi);
  t0 = (-1.f - oz) / (dz + 1e-15f); t1 = (1.f - oz) / (dz + 1e-15f);
  lo = fminf(t0, t1); hi = fmaxf(t0, t1); nv = fmaxf(nv, lo); fv = fminf(fv, hi);
  if (fv < nv) { nv = 1e9f; fv = 1e9f; }
  nv = fmaxf(nv, 0.05f);
  nearv = nv; farv = fv;
}

// Build the tile-invariant (per-lane) weight fragments once per kernel:
//   b1f[nt]: B 4x16 f32 fragment of W1 column-tile nt (row K=3 is the zero pad,
//            materialized as a zeroed 4th row of sW1 so the load is unconditional)
//   bias[nt]: b1 splat value for this lane's column
//   b2f[c2]: B 32x16 f16 fragment of W2 = [Ws | Wc | 0...] K-chunk c2
__device__ __forceinline__ void build_wfrags(int lm, int g,
                                             const float* sW1, const float* sb1,
                                             const _Float16* sW2,
                                             v2f b1f[4], float bias[4], v16h b2f[2]) {
  int k0 = 2 * g;
  #pragma unroll
  for (int nt = 0; nt < 4; ++nt) {
    int col = nt * 16 + lm;
    b1f[nt].x = sW1[k0 * HID + col];
    b1f[nt].y = sW1[(k0 + 1) * HID + col];
    bias[nt]  = sb1[col];
  }
  #pragma unroll
  for (int c2 = 0; c2 < 2; ++c2) {
    int kb = 32 * c2 + 8 * g;
    #pragma unroll
    for (int i = 0; i < 16; ++i)
      b2f[c2][i] = sW2[(kb + i + (i >= 8 ? 8 : 0)) * 16 + lm];
  }
}

// One 16-sample tile of the MLP:
//   layer 1: h = relu(A1[16x4] @ W1[4x64] + b1)   via 4x v_wmma_f32_16x16x4_f32
//   layer 2: O = h16[16x64] @ W2[64x16]           via 2x v_wmma_f32_16x16x32_f16
// Fragment layouts per ISA 7.12.2 (compile-only assumption):
//   A 16x4 f32 : lane<16 holds row M=lane, K={0,1}; lane>=16 holds K={2,3}
//   B 4x16 f32 : lane<16 holds col N=lane, K={0,1}; lane>=16 holds K={2,3}
//   A 16x32 f16: vgpr halves i=0..7 -> K = 8*g + i (+16 for i=8..15), row M=lane&15
//   B 32x16 f16: halves i=0..7 -> K = 8*g + i (+16 for i=8..15), col N=lane&15
//   C/D 16x16  : vgpr j, lane L -> row j + 8*(L>>4), col L&15
// The layer-1 D -> layer-2 A transpose bounces through LDS (DS ops are
// processed in-order within a wave; compiler inserts the s_wait_dscnt).
__device__ __forceinline__ v8f mlp_tile(v2f a1, int lm, int g,
                                        const v2f b1f[4], const float bias[4],
                                        const v16h b2f[2], _Float16* hb) {
  #pragma unroll
  for (int nt = 0; nt < 4; ++nt) {
    int col = nt * 16 + lm;
    v8f c;
    #pragma unroll
    for (int j = 0; j < 8; ++j) c[j] = bias[nt];
    c = __builtin_amdgcn_wmma_f32_16x16x4_f32(false, a1, false, b1f[nt],
                                              (short)0, c, false, false);
    #pragma unroll
    for (int j = 0; j < 8; ++j)  // relu + f16 pack, store h tile [16][64] to LDS
      hb[(j + 8 * g) * HID + col] = (_Float16)fmaxf(c[j], 0.f);
  }
  v8f o;
  #pragma unroll
  for (int j = 0; j < 8; ++j) o[j] = 0.f;
  #pragma unroll
  for (int c2 = 0; c2 < 2; ++c2) {
    v16h a2;
    int ka = 32 * c2 + 8 * g;
    #pragma unroll
    for (int i = 0; i < 8; ++i) {        // two 16B-contiguous runs per chunk
      a2[i]     = hb[lm * HID + ka + i];
      a2[8 + i] = hb[lm * HID + ka + 16 + i];
    }
    o = __builtin_amdgcn_wmma_f32_16x16x32_f16(false, a2, false, b2f[c2],
                                               (short)0, o, false, false);
  }
  return o;
}

// ---------------- Kernel A: coarse density sigma0[N,64] ----------------
__global__ __launch_bounds__(128) void k_coarse(
    const float* __restrict__ ro, const float* __restrict__ rd,
    const float* __restrict__ W1, const float* __restrict__ b1,
    const float* __restrict__ Ws, const float* __restrict__ bs,
    float* __restrict__ sigma0) {
  __shared__ float    sW1[4 * HID];  // row 3 = zeros (K pad)
  __shared__ float    sb1[HID];
  __shared__ _Float16 sW2[HID * 16];
  __shared__ _Float16 hbuf[4][16 * HID];

  int tid = threadIdx.x;
  for (int i = tid; i < 4 * HID; i += 128) sW1[i] = (i < 3 * HID) ? W1[i] : 0.f;
  for (int i = tid; i < HID; i += 128)     sb1[i] = b1[i];
  for (int i = tid; i < HID * 16; i += 128) {
    int k = i >> 4, n = i & 15;
    sW2[i] = (_Float16)((n == 0) ? Ws[k] : 0.f);  // only sigma column used here
  }
  __syncthreads();

  int wave = tid >> 5, lane = tid & 31, lm = lane & 15, g = lane >> 4;
  int r = blockIdx.x * 4 + wave;  // 4096 blocks * 4 waves = 16384 rays exactly
  float ox = ro[r * 3 + 0], oy = ro[r * 3 + 1], oz = ro[r * 3 + 2];
  float dx = rd[r * 3 + 0], dy = rd[r * 3 + 1], dz = rd[r * 3 + 2];
  float nearv, farv;
  ray_near_far(ox, oy, oz, dx, dy, dz, nearv, farv);
  float span = farv - nearv;
  float bsv = bs[0];
  _Float16* hb = hbuf[wave];

  v2f b1f[4]; float bias[4]; v16h b2f[2];
  build_wfrags(lm, g, sW1, sb1, sW2, b1f, bias, b2f);  // hoisted out of tile loop

  for (int tile = 0; tile < 4; ++tile) {
    int step = tile * 16 + lm;
    float z = nearv + span * (float)step * (1.f / 63.f);
    float px = clamp1(ox + dx * z), py = clamp1(oy + dy * z), pz = clamp1(oz + dz * z);
    v2f a1;
    a1.x = g ? pz : px;
    a1.y = g ? 0.f : py;  // K=3 zero pad
    v8f o = mlp_tile(a1, lm, g, b1f, bias, b2f, hb);
    if (lm == 0) {  // lanes 0,16 hold column 0 (sigma) for rows 0-7 / 8-15
      #pragma unroll
      for (int j = 0; j < 8; ++j)
        sigma0[r * NSTEP + tile * 16 + j + 8 * g] = softplus_f(o[j] + bsv);
    }
  }
}

// ------- Kernel B: importance sampling + merge -> z_all[N,128] ---------
__global__ __launch_bounds__(256) void k_sample(
    const float* __restrict__ ro, const float* __restrict__ rd,
    const float* __restrict__ sigma0, float* __restrict__ zall) {
  int r = blockIdx.x * 256 + threadIdx.x;
  if (r >= N_RAYS) return;
  float ox = ro[r * 3 + 0], oy = ro[r * 3 + 1], oz = ro[r * 3 + 2];
  float dx = rd[r * 3 + 0], dy = rd[r * 3 + 1], dz = rd[r * 3 + 2];
  float nearv, farv;
  ray_near_far(ox, oy, oz, dx, dy, dz, nearv, farv);
  float span = farv - nearv;
  float dz63 = span * (1.f / 63.f);  // uniform z-spacing of the coarse linspace

  // coarse weights w0[0..62] (only 1..62 feed the pdf)
  float w[63];
  float Tacc = 1.f;
  for (int i = 0; i < 63; ++i) {
    float a = 1.f - expf(-dz63 * sigma0[r * NSTEP + i]);
    w[i] = a * Tacc;
    Tacc *= (1.f - a + 1e-15f);
  }
  float s = 0.f;
  for (int i = 1; i < 63; ++i) s += w[i] + 1e-5f;
  float cdf[63];
  cdf[0] = 0.f;
  float acc = 0.f;
  for (int j = 1; j < 63; ++j) { acc += (w[j] + 1e-5f) / s; cdf[j] = acc; }

  // deterministic inverse-CDF: u ascending -> O(T) two-pointer searchsorted
  float nz[64];
  int p = 1;
  for (int k = 0; k < 64; ++k) {
    float u = (0.5f + (float)k) * (1.f / 64.f);
    while (p < 63 && cdf[p] <= u) ++p;  // searchsorted(side='right')
    int below = p - 1; if (below > 62) below = 62; if (below < 0) below = 0;
    int above = p;     if (above > 62) above = 62;
    float c0 = cdf[below], c1 = cdf[above];
    float b0 = nearv + span * ((float)below + 0.5f) * (1.f / 63.f);  // z_mid bins
    float b1v = nearv + span * ((float)above + 0.5f) * (1.f / 63.f);
    float den = c1 - c0; if (den < 1e-5f) den = 1.f;
    float tt = (u - c0) / den;
    nz[k] = b0 + tt * (b1v - b0);
  }

  // merge two sorted length-64 lists (stable: coarse first on ties)
  int i = 0, j = 0;
  for (int t = 0; t < TFINE; ++t) {
    float zv  = (i < 64) ? nearv + span * (float)i * (1.f / 63.f) : 3.4e38f;
    float nzv = (j < 64) ? nz[j] : 3.4e38f;
    if (zv <= nzv) { zall[r * TFINE + t] = zv;  ++i; }
    else           { zall[r * TFINE + t] = nzv; ++j; }
  }
}

// -------- Kernel C: full MLP on 128 samples/ray + compositing ----------
__global__ __launch_bounds__(128) void k_fine(
    const float* __restrict__ ro, const float* __restrict__ rd,
    const float* __restrict__ W1, const float* __restrict__ b1,
    const float* __restrict__ Ws, const float* __restrict__ bs,
    const float* __restrict__ Wc, const float* __restrict__ Wd,
    const float* __restrict__ bc, const float* __restrict__ zall,
    float* __restrict__ out) {
  __shared__ float    sW1[4 * HID];  // row 3 = zeros (K pad)
  __shared__ float    sb1[HID];
  __shared__ _Float16 sW2[HID * 16];        // [Ws | Wc | 0...]
  __shared__ _Float16 hbuf[4][16 * HID];
  __shared__ float    zb[4][TFINE];
  __shared__ float    sbf[4][TFINE];        // sigma per sample
  __shared__ float    rbf[4][TFINE * 4];    // rgb per sample

  int tid = threadIdx.x;
  for (int i = tid; i < 4 * HID; i += 128) sW1[i] = (i < 3 * HID) ? W1[i] : 0.f;
  for (int i = tid; i < HID; i += 128)     sb1[i] = b1[i];
  for (int i = tid; i < HID * 16; i += 128) {
    int k = i >> 4, n = i & 15;
    float v = (n == 0) ? Ws[k] : ((n < 5) ? Wc[k * 4 + (n - 1)] : 0.f);
    sW2[i] = (_Float16)v;
  }

  int wave = tid >> 5, lane = tid & 31, lm = lane & 15, g = lane >> 4;
  int r = blockIdx.x * 4 + wave;
  float ox = ro[r * 3 + 0], oy = ro[r * 3 + 1], oz = ro[r * 3 + 2];
  float dx = rd[r * 3 + 0], dy = rd[r * 3 + 1], dz = rd[r * 3 + 2];
  float nearv, farv;
  ray_near_far(ox, oy, oz, dx, dy, dz, nearv, farv);
  float span = farv - nearv;
  float sdist = span * (1.f / (float)NSTEP);  // sample_dist uses num_steps=64
  float bsv = bs[0];
  // d @ Wd + bc is a per-ray constant (dirs are broadcast along the ray)
  float dwd[4];
  #pragma unroll
  for (int c = 0; c < 4; ++c)
    dwd[c] = dx * Wd[0 * 4 + c] + dy * Wd[1 * 4 + c] + dz * Wd[2 * 4 + c] + bc[c];

  for (int t = lane; t < TFINE; t += 32) zb[wave][t] = zall[r * TFINE + t];
  __syncthreads();  // weights + per-wave z staged

  _Float16* hb = hbuf[wave];
  v2f b1f[4]; float bias[4]; v16h b2f[2];
  build_wfrags(lm, g, sW1, sb1, sW2, b1f, bias, b2f);  // hoisted out of tile loop

  for (int tile = 0; tile < 8; ++tile) {
    int step = tile * 16 + lm;
    float z = zb[wave][step];
    float px = clamp1(ox + dx * z), py = clamp1(oy + dy * z), pz = clamp1(oz + dz * z);
    v2f a1;
    a1.x = g ? pz : px;
    a1.y = g ? 0.f : py;
    v8f o = mlp_tile(a1, lm, g, b1f, bias, b2f, hb);
    if (lm < 5) {  // col 0 = sigma, cols 1..4 = rgb logits
      #pragma unroll
      for (int j = 0; j < 8; ++j) {
        int m = tile * 16 + j + 8 * g;
        float v = o[j];
        if (lm == 0) sbf[wave][m] = softplus_f(v + bsv);
        else         rbf[wave][m * 4 + (lm - 1)] = sigmoid_f(v + dwd[lm - 1]);
      }
    }
  }
  __syncthreads();  // uniform across block; make sigma/rgb visible

  if (lane == 0) {  // sequential 128-step composite per ray
    float Tacc = 1.f, wsum = 0.f, dep = 0.f;
    float img0 = 0.f, img1 = 0.f, img2 = 0.f, img3 = 0.f;
    float invspan = (span > 1e-12f) ? 1.f / span : 0.f;
    for (int t = 0; t < TFINE; ++t) {
      float delta = (t < TFINE - 1) ? (zb[wave][t + 1] - zb[wave][t]) : sdist;
      float alpha = 1.f - expf(-delta * sbf[wave][t]);
      float wv = alpha * Tacc;
      Tacc *= (1.f - alpha + 1e-15f);
      wsum += wv;
      float oznorm = fminf(fmaxf((zb[wave][t] - nearv) * invspan, 0.f), 1.f);
      dep += wv * oznorm;
      img0 += wv * rbf[wave][t * 4 + 0];
      img1 += wv * rbf[wave][t * 4 + 1];
      img2 += wv * rbf[wave][t * 4 + 2];
      img3 += wv * rbf[wave][t * 4 + 3];
    }
    float bg = 1.f - wsum;  // bg_color = 1
    out[r * 4 + 0] = img0 + bg;
    out[r * 4 + 1] = img1 + bg;
    out[r * 4 + 2] = img2 + bg;
    out[r * 4 + 3] = img3 + bg;
    out[N_RAYS * 4 + r] = 1.f - (dep + bg);   // depth
    out[N_RAYS * 5 + r] = wsum;               // weights_sum
  }
}

extern "C" void kernel_launch(void* const* d_in, const int* in_sizes, int n_in,
                              void* d_out, int out_size, void* d_ws, size_t ws_size,
                              hipStream_t stream) {
  const float* ro = (const float*)d_in[0];
  const float* rd = (const float*)d_in[1];
  const float* W1 = (const float*)d_in[2];
  const float* b1 = (const float*)d_in[3];
  const float* Ws = (const float*)d_in[4];
  const float* bs = (const float*)d_in[5];
  const float* Wc = (const float*)d_in[6];
  const float* Wd = (const float*)d_in[7];
  const float* bc = (const float*)d_in[8];
  // d_in[9]=num_steps(64), d_in[10]=upsample_steps(64): fixed by reference defaults.

  float* sigma0 = (float*)d_ws;                       // 16384*64 f32
  float* zall   = sigma0 + (size_t)N_RAYS * NSTEP;    // 16384*128 f32

  k_coarse<<<N_RAYS / 4, 128, 0, stream>>>(ro, rd, W1, b1, Ws, bs, sigma0);
  k_sample<<<N_RAYS / 256, 256, 0, stream>>>(ro, rd, sigma0, zall);
  k_fine<<<N_RAYS / 4, 128, 0, stream>>>(ro, rd, W1, b1, Ws, bs, Wc, Wd, bc,
                                         zall, (float*)d_out);
}
